// Encoder_43593918054694
// MI455X (gfx1250) — matrix operand order repaired and checked by reading, not compiled, not run on hardware
//
#include <hip/hip_runtime.h>
#include <hip/hip_bf16.h>
#include <cstdint>

// Bi-directional GRU encoder for MI455X (gfx1250, wave32, WMMA).
//   Phase 0: X -> bf16, W -> bf16 transposed [n][k]   (one-time, makes GEMM
//            operands half-size and LDS staging a pure 16B copy)
//   Phase 1: xg[d] = X @ W[d] + b[d][0]  (bf16 WMMA, fp32 acc, double-buffered
//            LDS staging via GLOBAL_LOAD_ASYNC_TO_LDS_B128 when available)
//   Phase 2: sequential GRU scan, U[d] held in registers as bf16 B-fragments,
//            h state in fp32 registers + bf16 LDS mirror for the WMMA A side.

typedef __attribute__((ext_vector_type(16))) __bf16 v16bf;
typedef __attribute__((ext_vector_type(8)))  float  v8f;

#define B_   64
#define S_   1024
#define E_   512
#define H2_  256
#define NG_  768          // 3*H2
#define MTOT (B_ * S_)    // 65536 rows

#if defined(__AMDGCN__) && __has_builtin(__builtin_amdgcn_global_load_async_to_lds_b128)
#define HAVE_ASYNC 1
#else
#define HAVE_ASYNC 0
#endif

#if HAVE_ASYNC
// Builtin parameter types (from hipcc diagnostic): pointer to GCC-vector
// 'int __attribute__((vector_size(16)))' in addrspace(1) / addrspace(3).
typedef int v4i_gcc __attribute__((vector_size(16)));
typedef __attribute__((address_space(1))) v4i_gcc gv4i_t;
typedef __attribute__((address_space(3))) v4i_gcc lv4i_t;
#endif

__device__ __forceinline__ float sigm(float x) { return 1.0f / (1.0f + __expf(-x)); }

__device__ __forceinline__ unsigned int pack2bf(float a, float b) {
    unsigned short ua = __builtin_bit_cast(unsigned short, (__bf16)a);
    unsigned short ub = __builtin_bit_cast(unsigned short, (__bf16)b);
    return (unsigned int)ua | ((unsigned int)ub << 16);
}

// 16-byte global -> LDS copy: async (ASYNCcnt-tracked) when available.
__device__ __forceinline__ void cp16(const __bf16* g, __bf16* l) {
#if HAVE_ASYNC
    __builtin_amdgcn_global_load_async_to_lds_b128(
        (gv4i_t*)(uintptr_t)g,
        (lv4i_t*)(unsigned int)(uintptr_t)l, 0, 0);
#else
    *(uint4*)l = *(const uint4*)g;
#endif
}

__device__ __forceinline__ void wait_async_le3() {
#if HAVE_ASYNC
    asm volatile("s_wait_asynccnt 0x3" ::: "memory");
#endif
}
__device__ __forceinline__ void wait_async_le0() {
#if HAVE_ASYNC
    asm volatile("s_wait_asynccnt 0x0" ::: "memory");
#endif
}

// ---------------------------------------------------------------------------
// Phase 0a: X fp32 -> bf16 (8 elems/thread, 16B stores)
// ---------------------------------------------------------------------------
__global__ __launch_bounds__(256)
void convert_x_kernel(const float* __restrict__ X, __bf16* __restrict__ Xb)
{
    size_t base = ((size_t)blockIdx.x * 256 + threadIdx.x) * 8;
    float4 v0 = *(const float4*)(X + base);
    float4 v1 = *(const float4*)(X + base + 4);
    uint4 o;
    o.x = pack2bf(v0.x, v0.y);
    o.y = pack2bf(v0.z, v0.w);
    o.z = pack2bf(v1.x, v1.y);
    o.w = pack2bf(v1.z, v1.w);
    *(uint4*)(Xb + base) = o;
}

// ---------------------------------------------------------------------------
// Phase 0b: W [k][n] fp32 -> Wt [n][k] bf16 (both directions via blockIdx.y)
// ---------------------------------------------------------------------------
__global__ __launch_bounds__(256)
void transpose_w_kernel(const float* __restrict__ Wf, const float* __restrict__ Wb,
                        __bf16* __restrict__ Wtf, __bf16* __restrict__ Wtb)
{
    const float* W  = (blockIdx.y == 0) ? Wf  : Wb;
    __bf16*      Wt = (blockIdx.y == 0) ? Wtf : Wtb;
    int idx = blockIdx.x * 256 + threadIdx.x;   // 0 .. E_*NG_-1
    int k = idx / NG_;
    int n = idx % NG_;
    Wt[(size_t)n * E_ + k] = (__bf16)W[idx];
}

// ---------------------------------------------------------------------------
// Phase 1: XG[d][m, n] = X[m, :] @ W[d] + bias[d][0][n]
// Tile: 64(M) x 128(N) x 32(K), 8 waves (2x4), each wave 32x32 (2x2 WMMA).
// Double-buffered LDS, staged with async global->LDS copies.
// ---------------------------------------------------------------------------
__global__ __launch_bounds__(256)
void proj_gemm_kernel(const __bf16* __restrict__ Xb,
                      const __bf16* __restrict__ Wtf, const float* __restrict__ bf,
                      const __bf16* __restrict__ Wtb, const float* __restrict__ bb,
                      float* __restrict__ XGf, float* __restrict__ XGb)
{
    const int d = blockIdx.z;
    const __bf16* Wt   = (d == 0) ? Wtf : Wtb;
    const float*  bias = (d == 0) ? bf  : bb;   // row 0 of [2, 768]
    float*        XG   = (d == 0) ? XGf : XGb;

    const int m0 = blockIdx.x * 64;
    const int n0 = blockIdx.y * 128;

    __shared__ __bf16 As[2][64 * 32];    // [m][k]
    __shared__ __bf16 Bs[2][128 * 32];   // [n][k]

    const int tid  = threadIdx.x;
    const int lane = tid & 31;
    const int wave = tid >> 5;
    const int wm   = wave & 1;
    const int wn   = wave >> 1;
    const int half = lane >> 4;
    const int l16  = lane & 15;

    // staging geometry: each thread copies 16B of A and 2x16B of B per K-tile
    const int srow = tid >> 2;            // 0..63
    const int skk  = (tid & 3) * 8;       // half offset within 32-wide K

    v8f c[2][2];
    #pragma unroll
    for (int tn = 0; tn < 2; ++tn) {
        float bv = bias[n0 + wn * 32 + tn * 16 + l16];
        #pragma unroll
        for (int tm = 0; tm < 2; ++tm)
            #pragma unroll
            for (int r = 0; r < 8; ++r) c[tm][tn][r] = bv;
    }

    auto issue = [&](int kt, int buf) {
        const int k0 = kt * 32;
        cp16(Xb + (size_t)(m0 + srow) * E_ + k0 + skk, &As[buf][srow * 32 + skk]);
        #pragma unroll
        for (int i = 0; i < 2; ++i) {
            int n = srow + i * 64;
            cp16(Wt + (size_t)(n0 + n) * E_ + k0 + skk, &Bs[buf][n * 32 + skk]);
        }
    };

    issue(0, 0);

    for (int kt = 0; kt < E_ / 32; ++kt) {
        const int cur = kt & 1;
        if (kt + 1 < E_ / 32) {
            issue(kt + 1, cur ^ 1);
            wait_async_le3();            // current tile's 3 copies complete
        } else {
            wait_async_le0();
        }
        __syncthreads();                 // all waves' staging visible

        // A frag (16x32): lanes 0-15 K {0..7,16..23}, lanes 16-31 K {8..15,24..31}
        v16bf a[2], b[2];
        #pragma unroll
        for (int tm = 0; tm < 2; ++tm) {
            const __bf16* p = &As[cur][(wm * 32 + tm * 16 + l16) * 32 + half * 8];
            ((uint4*)&a[tm])[0] = *(const uint4*)p;
            ((uint4*)&a[tm])[1] = *(const uint4*)(p + 16);
        }
        // B frag (32x16): lanes 0-15 K {0..7,16..23}, lanes 16-31 K {8..15,24..31}
        #pragma unroll
        for (int tn = 0; tn < 2; ++tn) {
            const __bf16* p = &Bs[cur][(wn * 32 + tn * 16 + l16) * 32 + half * 8];
            ((uint4*)&b[tn])[0] = *(const uint4*)p;
            ((uint4*)&b[tn])[1] = *(const uint4*)(p + 16);
        }
        #pragma unroll
        for (int tm = 0; tm < 2; ++tm)
            #pragma unroll
            for (int tn = 0; tn < 2; ++tn)
                c[tm][tn] = __builtin_amdgcn_wmma_f32_16x16x32_bf16(
                    false, a[tm], false, b[tn], (short)0, c[tm][tn], false, false);
        __syncthreads();                 // frag reads done before buffer reuse
    }

    #pragma unroll
    for (int tm = 0; tm < 2; ++tm)
        #pragma unroll
        for (int tn = 0; tn < 2; ++tn) {
            int ncol = n0 + wn * 32 + tn * 16 + l16;
            #pragma unroll
            for (int r = 0; r < 8; ++r) {
                int mrow = m0 + wm * 32 + tm * 16 + r + half * 8;
                XG[(size_t)mrow * NG_ + ncol] = c[tm][tn][r];
            }
        }
}

// ---------------------------------------------------------------------------
// Phase 2: masked GRU scan. One block = (direction, 16-row batch group).
// 768 threads = 24 waves; each wave owns 2 N-tiles of 16 cols (48 tiles = 768).
// ---------------------------------------------------------------------------
__global__ __launch_bounds__(768)
void gru_recur_kernel(const float* __restrict__ XGf, const float* __restrict__ XGb,
                      const float* __restrict__ Uf,  const float* __restrict__ bf,
                      const float* __restrict__ Ub,  const float* __restrict__ bb,
                      const float* __restrict__ h0f, const float* __restrict__ h0b,
                      const unsigned char* __restrict__ mask,
                      float* __restrict__ out)
{
    const int d  = blockIdx.y;
    const int bg = blockIdx.x;
    const float* XG   = (d == 0) ? XGf : XGb;
    const float* U    = (d == 0) ? Uf  : Ub;
    const float* bias = ((d == 0) ? bf : bb) + NG_;   // row 1
    const float* h0   = (d == 0) ? h0f : h0b;

    __shared__ __bf16 hB[16 * H2_];     // bf16 mirror of h (8 KB)
    __shared__ float  rec[16 * NG_];    // h @ U + b1 (48 KB)

    const int tid  = threadIdx.x;
    const int lane = tid & 31;
    const int wave = tid >> 5;          // 0..23
    const int half = lane >> 4;
    const int l16  = lane & 15;

    float hreg[6];
    #pragma unroll
    for (int it = 0; it < 6; ++it) {
        int idx = tid + it * 768;
        if (idx < 16 * H2_) {
            int m = idx >> 8, j = idx & 255;
            float v = h0[(size_t)(bg * 16 + m) * H2_ + j];
            hreg[it] = v;
            hB[idx]  = (__bf16)v;
        }
    }

    v16bf ub[2][8];
    float bias_v[2];
    #pragma unroll
    for (int t = 0; t < 2; ++t) {
        const int nb = (wave * 2 + t) * 16;
        bias_v[t] = bias[nb + l16];
        #pragma unroll
        for (int kb = 0; kb < 8; ++kb)
            #pragma unroll
            for (int i = 0; i < 16; ++i) {
                int koff = (i < 8) ? (i + 8 * half) : (i + 8 + 8 * half);
                ub[t][kb][i] = (__bf16)U[(size_t)(kb * 32 + koff) * NG_ + nb + l16];
            }
    }
    __syncthreads();

    for (int s = 0; s < S_; ++s) {
        const int sa = (d == 0) ? s : (S_ - 1 - s);

        v8f c[2];
        #pragma unroll
        for (int t = 0; t < 2; ++t)
            #pragma unroll
            for (int r = 0; r < 8; ++r) c[t][r] = bias_v[t];

        #pragma unroll
        for (int kb = 0; kb < 8; ++kb) {
            v16bf a;
            const __bf16* p = &hB[l16 * H2_ + kb * 32 + half * 8];
            ((uint4*)&a)[0] = *(const uint4*)p;
            ((uint4*)&a)[1] = *(const uint4*)(p + 16);
            #pragma unroll
            for (int t = 0; t < 2; ++t)
                c[t] = __builtin_amdgcn_wmma_f32_16x16x32_bf16(
                    false, a, false, ub[t][kb], (short)0, c[t], false, false);
        }
        #pragma unroll
        for (int t = 0; t < 2; ++t) {
            int n = (wave * 2 + t) * 16 + l16;
            #pragma unroll
            for (int r = 0; r < 8; ++r)
                rec[(r + 8 * half) * NG_ + n] = c[t][r];
        }
        __syncthreads();

        #pragma unroll
        for (int it = 0; it < 6; ++it) {
            int idx = tid + it * 768;
            if (idx < 16 * H2_) {
                int m = idx >> 8, j = idx & 255;
                int bglob = bg * 16 + m;
                size_t rowg = (size_t)bglob * S_ + sa;
                const float* xg = XG + rowg * NG_;
                float xz = xg[j], xr = xg[H2_ + j], xh = xg[2 * H2_ + j];
                float rz = rec[m * NG_ + j];
                float rr = rec[m * NG_ + H2_ + j];
                float rh = rec[m * NG_ + 2 * H2_ + j];
                float z  = sigm(xz + rz);
                float r  = sigm(xr + rr);
                float hh = tanhf(xh + r * rh);
                float hp = hreg[it];
                float hn = z * hp + (1.0f - z) * hh;
                hn = mask[rowg] ? hn : hp;
                hreg[it] = hn;
                hB[idx]  = (__bf16)hn;
                out[rowg * (2 * H2_) + (size_t)d * H2_ + j] = hn;
            }
        }
        __syncthreads();
    }

    #pragma unroll
    for (int it = 0; it < 6; ++it) {
        int idx = tid + it * 768;
        if (idx < 16 * H2_) {
            int m = idx >> 8, j = idx & 255;
            int bglob = bg * 16 + m;
            out[(size_t)B_ * S_ * (2 * H2_) + (size_t)bglob * (2 * H2_) + d * H2_ + j]
                = hreg[it];
        }
    }
}

// ---------------------------------------------------------------------------
extern "C" void kernel_launch(void* const* d_in, const int* in_sizes, int n_in,
                              void* d_out, int out_size, void* d_ws, size_t ws_size,
                              hipStream_t stream)
{
    // 0 tokens (unused), 1 embeddings, 2 mask(bool), 3 h0_f, 4 h0_b,
    // 5 Wf, 6 Uf, 7 bf, 8 Wb, 9 Ub, 10 bb
    const float*         X    = (const float*)d_in[1];
    const unsigned char* mask = (const unsigned char*)d_in[2];
    const float*         h0f  = (const float*)d_in[3];
    const float*         h0b  = (const float*)d_in[4];
    const float*         Wf   = (const float*)d_in[5];
    const float*         Uf   = (const float*)d_in[6];
    const float*         bf   = (const float*)d_in[7];
    const float*         Wb   = (const float*)d_in[8];
    const float*         Ub   = (const float*)d_in[9];
    const float*         bb   = (const float*)d_in[10];
    float* out = (float*)d_out;

    // workspace layout
    float*  XGf = (float*)d_ws;                           // [65536,768] f32
    float*  XGb = XGf + (size_t)MTOT * NG_;               // [65536,768] f32
    __bf16* Xb  = (__bf16*)(XGb + (size_t)MTOT * NG_);    // [65536,512] bf16
    __bf16* Wtf = Xb + (size_t)MTOT * E_;                 // [768,512]   bf16
    __bf16* Wtb = Wtf + (size_t)NG_ * E_;                 // [768,512]   bf16

    convert_x_kernel<<<(MTOT * E_) / (256 * 8), 256, 0, stream>>>(X, Xb);
    dim3 gT((E_ * NG_) / 256, 2);
    transpose_w_kernel<<<gT, 256, 0, stream>>>(Wf, Wb, Wtf, Wtb);

    dim3 g1(MTOT / 64, NG_ / 128, 2);
    proj_gemm_kernel<<<g1, 256, 0, stream>>>(Xb, Wtf, bf, Wtb, bb, XGf, XGb);

    dim3 g2(4, 2);
    gru_recur_kernel<<<g2, 768, 0, stream>>>(XGf, XGb, Uf, bf, Ub, bb,
                                             h0f, h0b, mask, out);
}
